// SlotAttention_39805756899706
// MI455X (gfx1250) — compile-verified, matrix operand order checked by reading, not run on hardware
//
#include <hip/hip_runtime.h>
#include <cmath>

#define B_   32
#define N_   1568
#define DIN_ 768
#define S_   8
#define D_   1536
#define H_   3072
#define EPS_    1e-8f
#define LN_EPS_ 1e-5f

typedef __bf16 bf16;
typedef __bf16 v16bf __attribute__((ext_vector_type(16)));
typedef float  v8f   __attribute__((ext_vector_type(8)));
typedef unsigned int v4u __attribute__((ext_vector_type(4)));
typedef long long i64;

// Load a 16-element bf16 fragment for one lane: two contiguous 16B chunks.
// With ptr pre-offset by (row*K + hi*8), chunk0 covers K 0..7(+hi*8), chunk1 covers K 16..23(+hi*8),
// matching the CDNA5 16-bit A/B fragment VGPR layout (CDNA5 ISA §7.12.2).
__device__ __forceinline__ v16bf load_frag(const bf16* p) {
  union { v4u u[2]; v16bf v; } t;
  t.u[0] = *reinterpret_cast<const v4u*>(p);
  t.u[1] = *reinterpret_cast<const v4u*>(p + 16);
  return t.v;
}

__device__ __forceinline__ v8f wmma_bf16(v16bf a, v16bf b, v8f c) {
  // (neg_a, A, neg_b, B, c_mod, C, reuse_a, reuse_b)
  return __builtin_amdgcn_wmma_f32_16x16x32_bf16(false, a, false, b, (short)0, c, false, false);
}

// Generic GEMM: C[M,N] = alpha * A[M,K] * Bt[N,K]^T   (row-major bf16 operands, f32 accum)
// Wave tile = (16*MT) x (16*NT). MODE 0: store f32 row-major, MODE 1: store bf16 row-major.
template<int MODE, int MT, int NT>
__global__ __launch_bounds__(256)
void wmma_gemm(const bf16* __restrict__ A, const bf16* __restrict__ Bt,
               void* __restrict__ Cout,
               int M, int Mvalid, int N, int K,
               i64 sA, i64 sB, i64 sC, int ldc, float alpha)
{
  const int b = blockIdx.y;
  const bf16* Ab = A + (i64)b * sA;
  const bf16* Bb = Bt + (i64)b * sB;

  const int wave = blockIdx.x * 8 + (threadIdx.x >> 5);
  const int lane = threadIdx.x & 31;
  const int tilesN = N / (16 * NT);
  const int tilesM = (M + 16 * MT - 1) / (16 * MT);
  if (wave >= tilesM * tilesN) return;
  const int tm = wave / tilesN, tn = wave % tilesN;
  const int m0 = tm * (16 * MT), n0 = tn * (16 * NT);

  const int r = lane & 15, hi = lane >> 4;

  const bf16* ap[MT];
#pragma unroll
  for (int mi = 0; mi < MT; ++mi) {
    int arow = m0 + mi * 16 + r;
    if (arow >= Mvalid) arow = Mvalid - 1;   // clamp; rows >= Mvalid never stored
    ap[mi] = Ab + (i64)arow * K + hi * 8;
  }
  const bf16* bp[NT];
#pragma unroll
  for (int ni = 0; ni < NT; ++ni)
    bp[ni] = Bb + (i64)(n0 + ni * 16 + r) * K + hi * 8;

  v8f acc[MT][NT] = {};

  for (int k0 = 0; k0 < K; k0 += 32) {
    if (k0 + 256 < K) {                      // global_prefetch_b8, 8 k-steps ahead
#pragma unroll
      for (int mi = 0; mi < MT; ++mi) __builtin_prefetch(ap[mi] + k0 + 256, 0, 1);
#pragma unroll
      for (int ni = 0; ni < NT; ++ni) __builtin_prefetch(bp[ni] + k0 + 256, 0, 1);
    }
    v16bf a[MT], bb[NT];
#pragma unroll
    for (int mi = 0; mi < MT; ++mi) a[mi] = load_frag(ap[mi] + k0);
#pragma unroll
    for (int ni = 0; ni < NT; ++ni) bb[ni] = load_frag(bp[ni] + k0);
#pragma unroll
    for (int mi = 0; mi < MT; ++mi)
#pragma unroll
      for (int ni = 0; ni < NT; ++ni)
        acc[mi][ni] = wmma_bf16(a[mi], bb[ni], acc[mi][ni]);
  }

#pragma unroll
  for (int mi = 0; mi < MT; ++mi) {
#pragma unroll
    for (int i = 0; i < 8; ++i) {
      const int row = m0 + mi * 16 + (hi << 3) + i;   // C/D layout: VGPR i -> M = i + 8*hi
      if (row >= Mvalid) continue;
#pragma unroll
      for (int ni = 0; ni < NT; ++ni) {
        const int col = n0 + ni * 16 + r;
        const float x = acc[mi][ni][i] * alpha;
        if (MODE == 0) {
          float* C = reinterpret_cast<float*>(Cout) + (i64)b * sC;
          C[(i64)row * ldc + col] = x;
        } else {
          bf16* C = reinterpret_cast<bf16*>(Cout) + (i64)b * sC;
          C[(i64)row * ldc + col] = (bf16)x;
        }
      }
    }
  }
}

__global__ __launch_bounds__(256)
void layernorm_to_bf16(const float* __restrict__ x, const float* __restrict__ g,
                       const float* __restrict__ bt, bf16* __restrict__ out, int Dd)
{
  __shared__ float s1[256], s2[256];
  const int row = blockIdx.x, tid = threadIdx.x;
  const float* xr = x + (i64)row * Dd;
  float a = 0.f, b = 0.f;
  for (int i = tid; i < Dd; i += 256) { float v = xr[i]; a += v; b += v * v; }
  s1[tid] = a; s2[tid] = b; __syncthreads();
  for (int off = 128; off > 0; off >>= 1) {
    if (tid < off) { s1[tid] += s1[tid + off]; s2[tid] += s2[tid + off]; }
    __syncthreads();
  }
  const float mean = s1[0] / (float)Dd;
  const float var  = s2[0] / (float)Dd - mean * mean;
  const float rs   = rsqrtf(var + LN_EPS_);
  bf16* op = out + (i64)row * Dd;
  for (int i = tid; i < Dd; i += 256)
    op[i] = (bf16)((xr[i] - mean) * rs * g[i] + bt[i]);
}

// softmax over the S (slot) axis for each (b, j), then + EPS
__global__ void slot_softmax(const float* __restrict__ dots, float* __restrict__ attn)
{
  int idx = blockIdx.x * 256 + threadIdx.x;
  if (idx >= B_ * N_) return;
  int b = idx / N_, j = idx - b * N_;
  const float* dp = dots + (i64)b * S_ * N_ + j;
  float v[S_]; float m = -3.4e38f;
#pragma unroll
  for (int i = 0; i < S_; ++i) { v[i] = dp[(i64)i * N_]; m = fmaxf(m, v[i]); }
  float s = 0.f;
#pragma unroll
  for (int i = 0; i < S_; ++i) { v[i] = __expf(v[i] - m); s += v[i]; }
  const float inv = 1.f / s;
  float* ap = attn + (i64)b * S_ * N_ + j;
#pragma unroll
  for (int i = 0; i < S_; ++i) ap[(i64)i * N_] = v[i] * inv + EPS_;
}

// renormalize each (b,i) row over j, emit bf16 for the updates GEMM
__global__ __launch_bounds__(256)
void attn_renorm(const float* __restrict__ attn, bf16* __restrict__ out)
{
  __shared__ float s1[256];
  const int row = blockIdx.x, tid = threadIdx.x;
  const float* ar = attn + (i64)row * N_;
  float s = 0.f;
  for (int j = tid; j < N_; j += 256) s += ar[j];
  s1[tid] = s; __syncthreads();
  for (int off = 128; off > 0; off >>= 1) {
    if (tid < off) s1[tid] += s1[tid + off];
    __syncthreads();
  }
  const float inv = 1.f / s1[0];
  bf16* op = out + (i64)row * N_;
  for (int j = tid; j < N_; j += 256) op[j] = (bf16)(ar[j] * inv);
}

__global__ void gru_elem(const float* __restrict__ gi, const float* __restrict__ gh,
                         const float* __restrict__ bih, const float* __restrict__ bhh,
                         const float* __restrict__ hprev, float* __restrict__ hnew)
{
  int idx = blockIdx.x * 256 + threadIdx.x;
  if (idx >= 256 * D_) return;
  const int row = idx / D_, d = idx - row * D_;
  const i64 base = (i64)row * 3 * D_;
  const float gr = gi[base + d]      + bih[d]      + gh[base + d]      + bhh[d];
  const float gz = gi[base + D_ + d] + bih[D_ + d] + gh[base + D_ + d] + bhh[D_ + d];
  const float r  = 1.f / (1.f + __expf(-gr));
  const float z  = 1.f / (1.f + __expf(-gz));
  const float gn = (gi[base + 2 * D_ + d] + bih[2 * D_ + d]) +
                   r * (gh[base + 2 * D_ + d] + bhh[2 * D_ + d]);
  const float n  = tanhf(gn);
  hnew[idx] = (1.f - z) * n + z * hprev[idx];
}

__global__ void f32_to_bf16_k(const float* __restrict__ in, bf16* __restrict__ out, int n)
{
  int i = blockIdx.x * 256 + threadIdx.x;
  if (i < n) out[i] = (bf16)in[i];
}

__global__ void relu_bias_bf16(const float* __restrict__ in, const float* __restrict__ bias,
                               bf16* __restrict__ out, int cols, int n)
{
  int i = blockIdx.x * 256 + threadIdx.x;
  if (i < n) {
    const int c = i % cols;
    out[i] = (bf16)fmaxf(in[i] + bias[c], 0.f);
  }
}

__global__ void add_bias_inplace(float* __restrict__ acc, const float* __restrict__ add,
                                 const float* __restrict__ bias, int cols, int n)
{
  int i = blockIdx.x * 256 + threadIdx.x;
  if (i < n) acc[i] += add[i] + bias[i % cols];
}

__global__ void init_slots_k(const float* __restrict__ init, float* __restrict__ slots)
{
  int i = blockIdx.x * 256 + threadIdx.x;
  if (i < 256 * D_) {
    const int row = i / D_, d = i - row * D_;
    slots[i] = init[(row & (S_ - 1)) * D_ + d];
  }
}

__global__ void copy_f32_k(const float* __restrict__ in, float* __restrict__ out, int n)
{
  int i = blockIdx.x * 256 + threadIdx.x;
  if (i < n) out[i] = in[i];
}

// LDS-tiled transpose+convert: f32 W[K,N] -> bf16 Wt[N,K]. K,N multiples of 64.
// 64x64 tile, pad to 65 columns to avoid LDS bank conflicts; both sides coalesced.
__global__ __launch_bounds__(256)
void transpose_to_bf16(const float* __restrict__ W, bf16* __restrict__ Wt, int K, int N)
{
  __shared__ float tile[64][65];
  const int nTilesN = N >> 6;
  const int bk = blockIdx.x / nTilesN, bn = blockIdx.x - bk * nTilesN;
  const int k0 = bk << 6, n0 = bn << 6;
  const int tx = threadIdx.x & 63, ty = threadIdx.x >> 6;   // ty in 0..3
#pragma unroll
  for (int rr = 0; rr < 16; ++rr) {
    const int kr = ty * 16 + rr;
    tile[kr][tx] = W[(i64)(k0 + kr) * N + n0 + tx];
  }
  __syncthreads();
#pragma unroll
  for (int rr = 0; rr < 16; ++rr) {
    const int nr = ty * 16 + rr;
    Wt[(i64)(n0 + nr) * K + k0 + tx] = (bf16)tile[tx][nr];
  }
}

extern "C" void kernel_launch(void* const* d_in, const int* in_sizes, int n_in,
                              void* d_out, int out_size, void* d_ws, size_t ws_size,
                              hipStream_t stream)
{
  (void)in_sizes; (void)n_in; (void)out_size; (void)ws_size;
  const float* inputs = (const float*)d_in[0];
  const float* init_s = (const float*)d_in[1];
  const float* Wq  = (const float*)d_in[2];
  const float* Wk  = (const float*)d_in[3];
  const float* Wv  = (const float*)d_in[4];
  const float* Wih = (const float*)d_in[5];
  const float* Whh = (const float*)d_in[6];
  const float* bih = (const float*)d_in[7];
  const float* bhh = (const float*)d_in[8];
  const float* W1  = (const float*)d_in[9];
  const float* b1  = (const float*)d_in[10];
  const float* W2  = (const float*)d_in[11];
  const float* b2  = (const float*)d_in[12];
  const float* g_in  = (const float*)d_in[13];
  const float* bt_in = (const float*)d_in[14];
  const float* g_s   = (const float*)d_in[15];
  const float* bt_s  = (const float*)d_in[16];
  const float* g_m   = (const float*)d_in[17];
  const float* bt_m  = (const float*)d_in[18];

  char* base = (char*)d_ws; size_t off = 0;
  auto alloc = [&](size_t bytes) -> void* {
    void* p = base + off; off = (off + bytes + 255) & ~(size_t)255; return p;
  };

  const int Mtok = B_ * N_;   // 50176 tokens
  bf16* xln    = (bf16*)alloc((size_t)Mtok * DIN_ * 2);
  bf16* kbf    = (bf16*)alloc((size_t)Mtok * D_ * 2);
  bf16* vT     = (bf16*)alloc((size_t)B_ * D_ * N_ * 2);   // [b][d][j]
  bf16* WqT    = (bf16*)alloc((size_t)D_ * D_ * 2);
  bf16* WkT    = (bf16*)alloc((size_t)D_ * DIN_ * 2);
  bf16* WvT    = (bf16*)alloc((size_t)D_ * DIN_ * 2);
  bf16* WihT   = (bf16*)alloc((size_t)3 * D_ * D_ * 2);
  bf16* WhhT   = (bf16*)alloc((size_t)3 * D_ * D_ * 2);
  bf16* W1T    = (bf16*)alloc((size_t)H_ * D_ * 2);
  bf16* W2T    = (bf16*)alloc((size_t)D_ * H_ * 2);
  float* slotsA = (float*)alloc((size_t)256 * D_ * 4);
  float* slotsB = (float*)alloc((size_t)256 * D_ * 4);
  bf16* sln    = (bf16*)alloc((size_t)256 * D_ * 2);
  bf16* qbf    = (bf16*)alloc((size_t)256 * D_ * 2);
  float* dots  = (float*)alloc((size_t)B_ * S_ * N_ * 4);
  float* attn  = (float*)alloc((size_t)B_ * S_ * N_ * 4);
  bf16* attnbf = (bf16*)alloc((size_t)B_ * S_ * N_ * 2);
  bf16* ubf    = (bf16*)alloc((size_t)256 * D_ * 2);
  bf16* slotsbf = (bf16*)alloc((size_t)256 * D_ * 2);
  float* gi    = (float*)alloc((size_t)256 * 3 * D_ * 4);
  float* gh    = (float*)alloc((size_t)256 * 3 * D_ * 4);
  bf16* pre    = (bf16*)alloc((size_t)256 * D_ * 2);
  float* hmid  = (float*)alloc((size_t)256 * H_ * 4);
  bf16* hrelu  = (bf16*)alloc((size_t)256 * H_ * 2);
  float* mlpo  = (float*)alloc((size_t)256 * D_ * 4);

  auto grid1 = [](int n) { return (n + 255) / 256; };
  auto ggrid = [](int M, int N, int MT, int NT) {
    int t = ((M + 16 * MT - 1) / (16 * MT)) * (N / (16 * NT));
    return (t + 7) / 8;
  };
  auto tgrid = [](int K, int N) { return (K >> 6) * (N >> 6); };

  // One-time weight convert+transpose to bf16 [N,K] (LDS-tiled, coalesced both sides)
  transpose_to_bf16<<<tgrid(D_, D_),     256, 0, stream>>>(Wq,  WqT,  D_,  D_);
  transpose_to_bf16<<<tgrid(DIN_, D_),   256, 0, stream>>>(Wk,  WkT,  DIN_, D_);
  transpose_to_bf16<<<tgrid(DIN_, D_),   256, 0, stream>>>(Wv,  WvT,  DIN_, D_);
  transpose_to_bf16<<<tgrid(D_, 3 * D_), 256, 0, stream>>>(Wih, WihT, D_,  3 * D_);
  transpose_to_bf16<<<tgrid(D_, 3 * D_), 256, 0, stream>>>(Whh, WhhT, D_,  3 * D_);
  transpose_to_bf16<<<tgrid(D_, H_),     256, 0, stream>>>(W1,  W1T,  D_,  H_);
  transpose_to_bf16<<<tgrid(H_, D_),     256, 0, stream>>>(W2,  W2T,  H_,  D_);

  // x = LN(inputs) -> bf16
  layernorm_to_bf16<<<Mtok, 256, 0, stream>>>(inputs, g_in, bt_in, xln, DIN_);

  // k = x @ Wk (bf16 [tok, D])
  wmma_gemm<1, 2, 4><<<dim3(ggrid(Mtok, D_, 2, 4), 1), 256, 0, stream>>>(
      xln, WkT, kbf, Mtok, Mtok, D_, DIN_, 0, 0, 0, D_, 1.0f);
  // vT[b][d][j] = (Wv^T @ x_b^T): A = WvT[D,DIN], Bt = xln_b[N,DIN]; coalesced bf16 stores
  wmma_gemm<1, 2, 2><<<dim3(ggrid(D_, N_, 2, 2), B_), 256, 0, stream>>>(
      WvT, xln, vT, D_, D_, N_, DIN_,
      0, (i64)N_ * DIN_, (i64)D_ * N_, N_, 1.0f);

  init_slots_k<<<grid1(256 * D_), 256, 0, stream>>>(init_s, slotsA);

  float* scur = slotsA; float* snxt = slotsB;
  const float scale = 1.0f / sqrtf((float)D_);

  for (int it = 0; it < 3; ++it) {
    // q = LN(slots) @ Wq  (bf16)
    layernorm_to_bf16<<<256, 256, 0, stream>>>(scur, g_s, bt_s, sln, D_);
    wmma_gemm<1, 2, 4><<<dim3(ggrid(256, D_, 2, 4), 1), 256, 0, stream>>>(
        sln, WqT, qbf, 256, 256, D_, D_, 0, 0, 0, D_, 1.0f);

    // dots[b] = scale * q_b[8,D] @ k_b[1568,D]^T   (batched, M padded to 16)
    wmma_gemm<0, 1, 2><<<dim3(ggrid(S_, N_, 1, 2), B_), 256, 0, stream>>>(
        qbf, kbf, dots, S_, S_, N_, D_,
        (i64)S_ * D_, (i64)N_ * D_, (i64)S_ * N_, N_, scale);

    slot_softmax<<<grid1(B_ * N_), 256, 0, stream>>>(dots, attn);
    attn_renorm<<<B_ * S_, 256, 0, stream>>>(attn, attnbf);

    // updates[b] = attn_b[8,1568] @ vT_b[D,1568]^T  (batched)
    wmma_gemm<1, 1, 4><<<dim3(ggrid(S_, D_, 1, 4), B_), 256, 0, stream>>>(
        attnbf, vT, ubf, S_, S_, D_, N_,
        (i64)S_ * N_, (i64)D_ * N_, (i64)S_ * D_, D_, 1.0f);

    // GRU gate GEMMs
    f32_to_bf16_k<<<grid1(256 * D_), 256, 0, stream>>>(scur, slotsbf, 256 * D_);
    wmma_gemm<0, 2, 4><<<dim3(ggrid(256, 3 * D_, 2, 4), 1), 256, 0, stream>>>(
        ubf, WihT, gi, 256, 256, 3 * D_, D_, 0, 0, 0, 3 * D_, 1.0f);
    wmma_gemm<0, 2, 4><<<dim3(ggrid(256, 3 * D_, 2, 4), 1), 256, 0, stream>>>(
        slotsbf, WhhT, gh, 256, 256, 3 * D_, D_, 0, 0, 0, 3 * D_, 1.0f);
    gru_elem<<<grid1(256 * D_), 256, 0, stream>>>(gi, gh, bih, bhh, scur, snxt);

    // MLP residual
    layernorm_to_bf16<<<256, 256, 0, stream>>>(snxt, g_m, bt_m, pre, D_);
    wmma_gemm<0, 2, 4><<<dim3(ggrid(256, H_, 2, 4), 1), 256, 0, stream>>>(
        pre, W1T, hmid, 256, 256, H_, D_, 0, 0, 0, H_, 1.0f);
    relu_bias_bf16<<<grid1(256 * H_), 256, 0, stream>>>(hmid, b1, hrelu, H_, 256 * H_);
    wmma_gemm<0, 2, 4><<<dim3(ggrid(256, D_, 2, 4), 1), 256, 0, stream>>>(
        hrelu, W2T, mlpo, 256, 256, D_, H_, 0, 0, 0, D_, 1.0f);
    add_bias_inplace<<<grid1(256 * D_), 256, 0, stream>>>(snxt, mlpo, b2, D_, 256 * D_);

    float* t = scur; scur = snxt; snxt = t;
  }

  copy_f32_k<<<grid1(256 * D_), 256, 0, stream>>>(scur, (float*)d_out, 256 * D_);
}
